// GNNEncoder_87153476370750
// MI455X (gfx1250) — matrix operand order repaired and checked by reading, not compiled, not run on hardware
//
#include <hip/hip_runtime.h>
#include <math.h>

typedef __attribute__((ext_vector_type(16))) __bf16 v16bf;
typedef __attribute__((ext_vector_type(8)))  float  v8f;

#define DIM 128
#define WAVES 4
#define BN_EPS 1e-5f

__device__ __forceinline__ float eluf(float x) {
    return x > 0.f ? x : (__expf(x) - 1.f);
}

__device__ __forceinline__ void atomAddF(float* p, float v) {
    __hip_atomic_fetch_add(p, v, __ATOMIC_RELAXED, __HIP_MEMORY_SCOPE_AGENT);
}

// ---- WMMA fragment access (CDNA5 ISA 7.12.2 layouts, wave32) ---------------

// A fragment: 16x32 bf16 tile from LDS, row-major with leading dim lda.
// lane&15 = row M; lanes 0-15 hold K offsets {0..7,16..23}, lanes 16-31 {8..15,24..31}
// v<4 -> K = kb+2v (+0,+1) contiguous 8 halfs; v>=4 -> K = 16+kb+... contiguous 8 halfs
// => compiler merges each group into a single ds_load_b128.
__device__ __forceinline__ v16bf load_a_frag(const __bf16* A, int lda, int k0, int lane) {
    int row = lane & 15;
    int kb  = (lane & 16) ? 8 : 0;
    v16bf a;
#pragma unroll
    for (int v = 0; v < 8; ++v) {
        int k = k0 + ((v < 4) ? (kb + 2 * v) : (16 + kb + 2 * (v - 4)));
        a[2 * v]     = A[row * lda + k];
        a[2 * v + 1] = A[row * lda + k + 1];
    }
    return a;
}

// B fragment from pre-swizzled bf16 weights: fragment-contiguous per lane,
// layout Wswz[kblk][ntile][lane][16] -> one 32B per-lane load (2x global_load_b128).
__device__ __forceinline__ v16bf load_b_swz(const __bf16* __restrict__ Wswz,
                                            int kblk, int t, int lane) {
    return *(const v16bf*)(Wswz + ((((kblk << 3) + t) << 5) + lane) * 16);
}

__device__ __forceinline__ v8f wmma_bf16(v16bf a, v16bf b, v8f c) {
    return __builtin_amdgcn_wmma_f32_16x16x32_bf16(false, a, false, b, (short)0, c,
                                                   false, false);
}

// ---- Kernels ---------------------------------------------------------------

__global__ void k_zero(float* p, long long n) {
    long long i = (long long)blockIdx.x * blockDim.x + threadIdx.x;
    long long stride = (long long)gridDim.x * blockDim.x;
    for (; i < n; i += stride) p[i] = 0.f;
}

// Convert f32 weight W[K][128] into WMMA B-fragment-swizzled bf16:
//   out[((kblk*8 + t)*32 + lane)*16 + j] = bf16(W[(kblk*32 + ((lane&16)?8:0) + (j&7) + ((j&8)?16:0))*128
//                                                 + t*16 + (lane&15)])
__global__ void k_swz_weight(const float* __restrict__ W, int K, __bf16* __restrict__ out) {
    int total = K * DIM;  // == (K/32)*8*32*16
    int i = blockIdx.x * blockDim.x + threadIdx.x;
    int stride = gridDim.x * blockDim.x;
    for (; i < total; i += stride) {
        int j    = i & 15;
        int lane = (i >> 4) & 31;
        int t    = (i >> 9) & 7;
        int kblk = i >> 12;
        int k = kblk * 32 + ((lane & 16) ? 8 : 0) + (j & 7) + ((j & 8) ? 16 : 0);
        int n = t * 16 + (lane & 15);
        out[i] = (__bf16)W[k * DIM + n];
    }
}

// One wave per 'responds' edge: scatter x_student[src] -> agg_item[dst],
// x_item[dst] -> agg_stu[src]; count both directions.
__global__ void k_scatter_responds(const float* __restrict__ xs, const float* __restrict__ xi,
                                   const int* __restrict__ src, const int* __restrict__ dst,
                                   float* agg_item, float* cnt_item,
                                   float* agg_stu, float* cnt_stu, int E) {
    int lane = threadIdx.x & 31;
    int e = blockIdx.x * (blockDim.x >> 5) + (threadIdx.x >> 5);
    if (e >= E) return;
    int s = src[e], d = dst[e];
    float4 vs = ((const float4*)(xs + (size_t)s * DIM))[lane];
    float4 vd = ((const float4*)(xi + (size_t)d * DIM))[lane];
    float* ai = agg_item + (size_t)d * DIM + lane * 4;
    float* as = agg_stu  + (size_t)s * DIM + lane * 4;
    atomAddF(ai + 0, vs.x); atomAddF(ai + 1, vs.y);
    atomAddF(ai + 2, vs.z); atomAddF(ai + 3, vs.w);
    atomAddF(as + 0, vd.x); atomAddF(as + 1, vd.y);
    atomAddF(as + 2, vd.z); atomAddF(as + 3, vd.w);
    if (lane == 0) { atomAddF(cnt_item + d, 1.f); atomAddF(cnt_stu + s, 1.f); }
}

// SimpleConv edge MLP, one wave per 16-edge tile:
//   tmp[16,256] = [x_j, x_i-x_j]; h = elu(tmp@W1+b1); msg = h@W2+b2 -> scatter-add at dst
__global__ void k_edge_mlp(const float* __restrict__ x,
                           const int* __restrict__ src, const int* __restrict__ dst,
                           const __bf16* __restrict__ W1s, const float* __restrict__ b1,
                           const __bf16* __restrict__ W2s, const float* __restrict__ b2,
                           float* sum_simple, float* cnt_p, int E) {
    __shared__ __bf16 smem[WAVES * (16 * 256 + 16 * DIM)];
    int lane = threadIdx.x & 31;
    int wave = threadIdx.x >> 5;
    int tile = blockIdx.x * WAVES + wave;
    int e0 = tile * 16;
    __bf16* tmp  = smem + wave * (16 * 256 + 16 * DIM);
    __bf16* hbuf = tmp + 16 * 256;

    for (int idx = lane; idx < 16 * DIM; idx += 32) {
        int r = idx >> 7, c = idx & (DIM - 1);
        int e = e0 + r;
        float xj = 0.f, xd = 0.f;
        if (e < E) {
            xj = x[(size_t)src[e] * DIM + c];
            xd = x[(size_t)dst[e] * DIM + c];
        }
        tmp[r * 256 + c]       = (__bf16)xj;
        tmp[r * 256 + DIM + c] = (__bf16)(xd - xj);
    }
    __syncthreads();

    int col = lane & 15;
    int mhi = (lane >> 4) << 3;

    // stage 1: h = elu(tmp @ W1 + b1), K = 256
    for (int t = 0; t < 8; ++t) {
        float bias = b1[t * 16 + col];
        v8f acc;
#pragma unroll
        for (int r = 0; r < 8; ++r) acc[r] = bias;
#pragma unroll
        for (int ks = 0; ks < 8; ++ks) {
            v16bf a = load_a_frag(tmp, 256, ks * 32, lane);
            v16bf b = load_b_swz(W1s, ks, t, lane);
            acc = wmma_bf16(a, b, acc);
        }
#pragma unroll
        for (int r = 0; r < 8; ++r)
            hbuf[(mhi + r) * DIM + t * 16 + col] = (__bf16)eluf(acc[r]);
    }
    __syncthreads();

    // stage 2: msg = h @ W2 + b2, K = 128; scatter to dst
    for (int t = 0; t < 8; ++t) {
        float bias = b2[t * 16 + col];
        v8f acc;
#pragma unroll
        for (int r = 0; r < 8; ++r) acc[r] = bias;
#pragma unroll
        for (int ks = 0; ks < 4; ++ks) {
            v16bf a = load_a_frag(hbuf, DIM, ks * 32, lane);
            v16bf b = load_b_swz(W2s, ks, t, lane);
            acc = wmma_bf16(a, b, acc);
        }
#pragma unroll
        for (int r = 0; r < 8; ++r) {
            int e = e0 + mhi + r;
            if (e < E) atomAddF(&sum_simple[(size_t)dst[e] * DIM + t * 16 + col], acc[r]);
        }
    }
    if (lane < 16) {
        int e = e0 + lane;
        if (e < E) atomAddF(&cnt_p[dst[e]], 1.f);
    }
}

// SAGE node GEMM, one wave per 16-row tile:
//   out = elu( (agg/cnt)@Wl + b + x@Wr )                             (item path)
//   out = elu( 0.5*((agg/cnt)@Wl + b + x@Wr + sum_simple/cnt_p) )    (student path)
__global__ void k_sage(const float* __restrict__ agg, const float* __restrict__ cnt,
                       const float* __restrict__ xv,
                       const __bf16* __restrict__ Wls, const __bf16* __restrict__ Wrs,
                       const float* __restrict__ bias,
                       const float* __restrict__ sum_simple, const float* __restrict__ cnt_p,
                       float* out, int N) {
    __shared__ __bf16 smem[WAVES * (2 * 16 * DIM)];
    int lane = threadIdx.x & 31, wave = threadIdx.x >> 5;
    int tile = blockIdx.x * WAVES + wave;
    int r0 = tile * 16;
    __bf16* Am = smem + wave * (2 * 16 * DIM);
    __bf16* Ax = Am + 16 * DIM;

    for (int idx = lane; idx < 16 * DIM; idx += 32) {
        int r = idx >> 7, c = idx & (DIM - 1);
        int row = r0 + r;
        float mv = 0.f, xvv = 0.f;
        if (row < N) {
            float cn = cnt[row]; cn = cn > 1.f ? cn : 1.f;
            mv  = agg[(size_t)row * DIM + c] / cn;
            xvv = xv[(size_t)row * DIM + c];
        }
        Am[r * DIM + c] = (__bf16)mv;
        Ax[r * DIM + c] = (__bf16)xvv;
    }
    __syncthreads();

    int col = lane & 15, mhi = (lane >> 4) << 3;
    for (int t = 0; t < 8; ++t) {
        float bv = bias[t * 16 + col];
        v8f acc;
#pragma unroll
        for (int r = 0; r < 8; ++r) acc[r] = bv;
#pragma unroll
        for (int ks = 0; ks < 4; ++ks) {
            acc = wmma_bf16(load_a_frag(Am, DIM, ks * 32, lane),
                            load_b_swz(Wls, ks, t, lane), acc);
            acc = wmma_bf16(load_a_frag(Ax, DIM, ks * 32, lane),
                            load_b_swz(Wrs, ks, t, lane), acc);
        }
#pragma unroll
        for (int r = 0; r < 8; ++r) {
            int row = r0 + mhi + r;
            if (row < N) {
                float v = acc[r];
                if (sum_simple) {
                    float cp = cnt_p[row]; cp = cp > 1.f ? cp : 1.f;
                    v = 0.5f * (v + sum_simple[(size_t)row * DIM + t * 16 + col] / cp);
                }
                out[(size_t)row * DIM + t * 16 + col] = eluf(v);
            }
        }
    }
}

// Column sums / sums-of-squares (thread t owns column t; one atomic per block-column)
__global__ void k_bn_stats(const float* __restrict__ X, int N, float* sum, float* sumsq) {
    int c = threadIdx.x;  // blockDim.x == 128
    float s = 0.f, q = 0.f;
    for (int r = blockIdx.x; r < N; r += gridDim.x) {
        float v = X[(size_t)r * DIM + c];
        s += v; q += v * v;
    }
    atomAddF(&sum[c], s);
    atomAddF(&sumsq[c], q);
}

__global__ void k_bn_norm(float* X, int N, const float* __restrict__ sum,
                          const float* __restrict__ sumsq,
                          const float* __restrict__ g, const float* __restrict__ b) {
    long long total = (long long)N * DIM;
    long long i = (long long)blockIdx.x * blockDim.x + threadIdx.x;
    long long stride = (long long)gridDim.x * blockDim.x;
    float invN = 1.f / (float)N;
    for (; i < total; i += stride) {
        int c = (int)(i & (DIM - 1));
        float mu  = sum[c] * invN;
        float var = sumsq[c] * invN - mu * mu;
        float v = X[i];
        X[i] = g[c] * (v - mu) * rsqrtf(var + BN_EPS) + b[c];
    }
}

// ---- Host launch -----------------------------------------------------------

extern "C" void kernel_launch(void* const* d_in, const int* in_sizes, int n_in,
                              void* d_out, int out_size, void* d_ws, size_t ws_size,
                              hipStream_t stream) {
    (void)n_in; (void)out_size; (void)ws_size;
    const float* x_student = (const float*)d_in[0];
    const float* x_item    = (const float*)d_in[1];
    const int*   r_src     = (const int*)d_in[2];
    const int*   r_dst     = (const int*)d_in[3];
    const int*   p_src     = (const int*)d_in[4];
    const int*   p_dst     = (const int*)d_in[5];
    const float* Wl_it = (const float*)d_in[6];
    const float* Wr_it = (const float*)d_in[7];
    const float* b_it  = (const float*)d_in[8];
    const float* Wl_st = (const float*)d_in[9];
    const float* Wr_st = (const float*)d_in[10];
    const float* b_st  = (const float*)d_in[11];
    const float* W1    = (const float*)d_in[12];
    const float* b1    = (const float*)d_in[13];
    const float* W2    = (const float*)d_in[14];
    const float* b2    = (const float*)d_in[15];
    const float* g_it  = (const float*)d_in[16];
    const float* be_it = (const float*)d_in[17];
    const float* g_st  = (const float*)d_in[18];
    const float* be_st = (const float*)d_in[19];

    const int d      = DIM;
    const int n_stu  = in_sizes[0] / d;
    const int n_item = in_sizes[1] / d;
    const int e_r    = in_sizes[2];
    const int e_p    = in_sizes[4];

    // ---- workspace layout: f32 accumulation scratch, then swizzled bf16 weights
    float* ws = (float*)d_ws;
    size_t off = 0;
    float* agg_item   = ws + off; off += (size_t)n_item * d;
    float* cnt_item   = ws + off; off += n_item;
    float* agg_stu    = ws + off; off += (size_t)n_stu * d;
    float* cnt_stu    = ws + off; off += n_stu;
    float* sum_simple = ws + off; off += (size_t)n_stu * d;
    float* cnt_p      = ws + off; off += n_stu;
    float* bn_sum_it  = ws + off; off += d;
    float* bn_sq_it   = ws + off; off += d;
    float* bn_sum_st  = ws + off; off += d;
    float* bn_sq_st   = ws + off; off += d;
    size_t zero_floats = off;

    off = (off + 15) & ~(size_t)15;  // 64B align for b128 fragment loads
    __bf16* bw = (__bf16*)(ws + off);
    size_t boff = 0;
    __bf16* W1s    = bw + boff; boff += (size_t)2 * d * d;  // [256][128]
    __bf16* W2s    = bw + boff; boff += (size_t)d * d;
    __bf16* Wl_its = bw + boff; boff += (size_t)d * d;
    __bf16* Wr_its = bw + boff; boff += (size_t)d * d;
    __bf16* Wl_sts = bw + boff; boff += (size_t)d * d;
    __bf16* Wr_sts = bw + boff; boff += (size_t)d * d;

    float* out_item = (float*)d_out;
    float* out_stu  = out_item + (size_t)n_item * d;

    // 1. zero all accumulation scratch
    k_zero<<<2048, 256, 0, stream>>>(ws, (long long)zero_floats);

    // 2. pre-swizzle weights to WMMA B-fragment bf16 layout (tiny, once per launch)
    k_swz_weight<<<64, 256, 0, stream>>>(W1,    2 * d, W1s);
    k_swz_weight<<<32, 256, 0, stream>>>(W2,    d,     W2s);
    k_swz_weight<<<32, 256, 0, stream>>>(Wl_it, d,     Wl_its);
    k_swz_weight<<<32, 256, 0, stream>>>(Wr_it, d,     Wr_its);
    k_swz_weight<<<32, 256, 0, stream>>>(Wl_st, d,     Wl_sts);
    k_swz_weight<<<32, 256, 0, stream>>>(Wr_st, d,     Wr_sts);

    // 3. SAGE mean-aggregation scatters over 'responds' (both directions)
    k_scatter_responds<<<(e_r + 7) / 8, 256, 0, stream>>>(
        x_student, x_item, r_src, r_dst, agg_item, cnt_item, agg_stu, cnt_stu, e_r);

    // 4. SimpleConv per-edge MLP over 'preceeds' (WMMA)
    int tiles_p = (e_p + 15) / 16;
    k_edge_mlp<<<(tiles_p + WAVES - 1) / WAVES, 32 * WAVES, 0, stream>>>(
        x_student, p_src, p_dst, W1s, b1, W2s, b2, sum_simple, cnt_p, e_p);

    // 5. item SAGE GEMM + elu -> d_out (WMMA)
    int tiles_i = (n_item + 15) / 16;
    k_sage<<<(tiles_i + WAVES - 1) / WAVES, 32 * WAVES, 0, stream>>>(
        agg_item, cnt_item, x_item, Wl_its, Wr_its, b_it, nullptr, nullptr, out_item, n_item);

    // 6. student SAGE GEMM fused with SimpleConv mean-combine + elu -> d_out (WMMA)
    int tiles_s = (n_stu + 15) / 16;
    k_sage<<<(tiles_s + WAVES - 1) / WAVES, 32 * WAVES, 0, stream>>>(
        agg_stu, cnt_stu, x_student, Wl_sts, Wr_sts, b_st, sum_simple, cnt_p, out_stu, n_stu);

    // 7. BatchNorm: stats then in-place normalize
    k_bn_stats<<<512, 128, 0, stream>>>(out_item, n_item, bn_sum_it, bn_sq_it);
    k_bn_stats<<<512, 128, 0, stream>>>(out_stu,  n_stu,  bn_sum_st, bn_sq_st);
    k_bn_norm<<<4096, 256, 0, stream>>>(out_item, n_item, bn_sum_it, bn_sq_it, g_it, be_it);
    k_bn_norm<<<4096, 256, 0, stream>>>(out_stu,  n_stu,  bn_sum_st, bn_sq_st, g_st, be_st);
}
